// Model_74448963109571
// MI455X (gfx1250) — matrix operand order repaired and checked by reading, not compiled, not run on hardware
//
#include <hip/hip_runtime.h>
#include <hip/hip_bf16.h>
#include <math.h>

typedef __attribute__((ext_vector_type(16))) _Float16 v16h;
typedef __attribute__((ext_vector_type(8)))  _Float16 v8h;
typedef __attribute__((ext_vector_type(8)))  float    v8f;

// Dimensions (fixed by the reference)
#define SEQ    4096
#define BATCH  8
#define INPUT  1024
#define HIDDEN 256
#define OUTPUT 128
#define MROWS  (SEQ*BATCH)   // 32768
// Padded LDS row stride for the scan's h buffer: 264 f16 = 132 dwords,
// 132 % 64 = 4  ->  lane lm starts at bank 4*lm, b128 covers banks
// 4lm..4lm+3  ->  conflict-free A-fragment loads.
#define HSTRIDE 264

__device__ __forceinline__ float fast_tanh(float x) {
#if __has_builtin(__builtin_amdgcn_tanhf)
    return __builtin_amdgcn_tanhf(x);
#elif __has_builtin(__builtin_amdgcn_tanh_f32)
    return __builtin_amdgcn_tanh_f32(x);
#else
    return tanhf(x);
#endif
}

// ---------------------------------------------------------------------------
// Convert fp32 weight (K x N, row-major) to f16 pre-swizzled into WMMA
// B-fragment order: out[((nt*(K/32))+kc)*512 + L*16 + e] where
//   K-index = kc*32 + (L>=16 ? 8 : 0) + e + (e>=8 ? 8 : 0) ,  N-index = nt*16 + (L&15)
// so a wave's B fragment is one contiguous 32-byte load per lane.
// kcShift = log2(K/32)  (always a power of two here).
// ---------------------------------------------------------------------------
__global__ void rnn_swizzle_w(const float* __restrict__ W,
                              _Float16* __restrict__ out, int K, int N,
                              int kcShift) {
    int total = K * N;
    int kcMask = (1 << kcShift) - 1;
    for (int i = blockIdx.x * blockDim.x + threadIdx.x; i < total;
         i += gridDim.x * blockDim.x) {
        int pos = i & 511;
        int tmp = i >> 9;
        int kc  = tmp & kcMask;
        int nt  = tmp >> kcShift;
        int L   = pos >> 4;
        int e   = pos & 15;
        int k   = kc * 32 + ((L & 16) >> 1) + e + (e & 8);
        int n   = nt * 16 + (L & 15);
        out[i] = (_Float16)W[(size_t)k * N + n];
    }
}

// ---------------------------------------------------------------------------
// Kernel 1: x_proj = x @ W_xh + b_i2h   (M=32768, K=1024, N=256), f32 out.
// One wave per 16x16 tile; 4 waves per block (4 M-tiles, shared N-tile).
// ---------------------------------------------------------------------------
__global__ __launch_bounds__(128) void rnn_gemm_xproj(
    const float* __restrict__ x, const _Float16* __restrict__ WxhSw,
    const float* __restrict__ bias, float* __restrict__ xp) {
    int wave = threadIdx.x >> 5, lane = threadIdx.x & 31;
    int lm = lane & 15, hio = (lane >> 4) & 1;
    int mt = blockIdx.x * 4 + wave;   // 0..2047
    int nt = blockIdx.y;              // 0..15

    const float* arow = x + (size_t)(mt * 16 + lm) * INPUT;
    const _Float16* bbase = WxhSw + (size_t)nt * (INPUT / 32) * 512 + lane * 16;

    v8f c = {};
#pragma unroll 4
    for (int kc = 0; kc < INPUT / 32; ++kc) {
        int k0 = kc * 32 + hio * 8;
        float4 p0 = *(const float4*)(arow + k0);
        float4 p1 = *(const float4*)(arow + k0 + 4);
        float4 p2 = *(const float4*)(arow + k0 + 16);
        float4 p3 = *(const float4*)(arow + k0 + 20);
        float f[16] = {p0.x, p0.y, p0.z, p0.w, p1.x, p1.y, p1.z, p1.w,
                       p2.x, p2.y, p2.z, p2.w, p3.x, p3.y, p3.z, p3.w};
        v16h a;
#pragma unroll
        for (int e = 0; e < 16; ++e) a[e] = (_Float16)f[e];
        v16h b = *(const v16h*)(bbase + (size_t)kc * 512);
        c = __builtin_amdgcn_wmma_f32_16x16x32_f16(false, a, false, b,
                                                   (short)0, c, false, false);
    }
    int n = nt * 16 + lm;
    float bv = bias[n];
#pragma unroll
    for (int v = 0; v < 8; ++v) {
        int Mr = mt * 16 + hio * 8 + v;
        xp[(size_t)Mr * HIDDEN + n] = c[v] + bv;
    }
}

// ---------------------------------------------------------------------------
// Kernel 2: sequential scan. One workgroup, 16 waves (512 threads).
// Wave w owns hidden columns [16w, 16w+16). W_hh B-fragments resident in
// VGPRs; h double-buffered in LDS (16 x HSTRIDE f16, rows 8..15 zero pad,
// row stride padded for bank-conflict-free ds_load_b128).
// Each step: 8 WMMAs (K=256) + v_tanh + writeback + xp prefetch + 1 barrier.
// ---------------------------------------------------------------------------
__global__ __launch_bounds__(512) void rnn_scan(
    const float* __restrict__ xp, const _Float16* __restrict__ WhhSw,
    const float* __restrict__ h0, _Float16* __restrict__ hs) {
    __shared__ _Float16 hbuf[2][16 * HSTRIDE];
    int tid = threadIdx.x;
    int w = tid >> 5, lane = tid & 31, lm = lane & 15, hio = (lane >> 4) & 1;

    // init: buffer0 = [h0 ; 0-pad], buffer1 = 0 (pad rows never rewritten)
    for (int i = tid; i < 16 * HSTRIDE; i += 512) {
        int row = i / HSTRIDE;
        int cp  = i - row * HSTRIDE;
        float v = (row < BATCH && cp < HIDDEN) ? h0[row * HIDDEN + cp] : 0.f;
        hbuf[0][i] = (_Float16)v;
        hbuf[1][i] = (_Float16)0.f;
    }

    // resident W_hh B fragments (8 k-chunks x 8 VGPRs = 64 VGPRs)
    v16h Bf[8];
#pragma unroll
    for (int kc = 0; kc < 8; ++kc)
        Bf[kc] = *(const v16h*)(WhhSw + ((size_t)w * 8 + kc) * 512 + lane * 16);

    int col = 16 * w + lm;
    // xp registers for t=0 (lanes 16..31 are padding rows -> 0)
    float xpr[8];
#pragma unroll
    for (int v = 0; v < 8; ++v)
        xpr[v] = (lane < 16) ? xp[(size_t)v * HIDDEN + col] : 0.f;

    __syncthreads();

    for (int t = 0; t < SEQ; ++t) {
        const _Float16* cur = hbuf[t & 1];
        _Float16* nxt = hbuf[(t + 1) & 1];

        v8f c = {};
#pragma unroll
        for (int kc = 0; kc < 8; ++kc) {
            int k0 = kc * 32 + hio * 8;
            const _Float16* arow = cur + lm * HSTRIDE + k0;
            v8h lo = *(const v8h*)(arow);
            v8h hh = *(const v8h*)(arow + 16);
            v16h a = __builtin_shufflevector(lo, hh, 0, 1, 2, 3, 4, 5, 6, 7,
                                             8, 9, 10, 11, 12, 13, 14, 15);
            c = __builtin_amdgcn_wmma_f32_16x16x32_f16(
                false, a, false, Bf[kc], (short)0, c, false, false);
        }

        if (lane < 16) {
#pragma unroll
            for (int v = 0; v < 8; ++v) {
                float hv = fast_tanh(c[v] + xpr[v]);
                _Float16 h16 = (_Float16)hv;
                nxt[v * HSTRIDE + col] = h16;                      // LDS
                hs[((size_t)t * BATCH + v) * HIDDEN + col] = h16;  // global f16
            }
            if (t + 1 < SEQ) {
#pragma unroll
                for (int v = 0; v < 8; ++v)
                    xpr[v] = xp[((size_t)(t + 1) * BATCH + v) * HIDDEN + col];
            }
        }
        __syncthreads();
    }
}

// ---------------------------------------------------------------------------
// Kernel 3: out = hs @ W_h2o + b_h2o   (M=32768, K=256, N=128), f32 out.
// ---------------------------------------------------------------------------
__global__ __launch_bounds__(128) void rnn_gemm_out(
    const _Float16* __restrict__ hs, const _Float16* __restrict__ Wh2oSw,
    const float* __restrict__ bias, float* __restrict__ out) {
    int wave = threadIdx.x >> 5, lane = threadIdx.x & 31;
    int lm = lane & 15, hio = (lane >> 4) & 1;
    int mt = blockIdx.x * 4 + wave;   // 0..2047
    int nt = blockIdx.y;              // 0..7

    const _Float16* arow = hs + (size_t)(mt * 16 + lm) * HIDDEN;
    const _Float16* bbase = Wh2oSw + (size_t)nt * (HIDDEN / 32) * 512 + lane * 16;

    v8f c = {};
#pragma unroll
    for (int kc = 0; kc < HIDDEN / 32; ++kc) {
        int k0 = kc * 32 + hio * 8;
        v8h lo = *(const v8h*)(arow + k0);
        v8h hh = *(const v8h*)(arow + k0 + 16);
        v16h a = __builtin_shufflevector(lo, hh, 0, 1, 2, 3, 4, 5, 6, 7,
                                         8, 9, 10, 11, 12, 13, 14, 15);
        v16h b = *(const v16h*)(bbase + (size_t)kc * 512);
        c = __builtin_amdgcn_wmma_f32_16x16x32_f16(false, a, false, b,
                                                   (short)0, c, false, false);
    }
    int n = nt * 16 + lm;
    float bv = bias[n];
#pragma unroll
    for (int v = 0; v < 8; ++v) {
        int Mr = mt * 16 + hio * 8 + v;
        out[(size_t)Mr * OUTPUT + n] = c[v] + bv;
    }
}

// ---------------------------------------------------------------------------
// Launch
// ---------------------------------------------------------------------------
extern "C" void kernel_launch(void* const* d_in, const int* in_sizes, int n_in,
                              void* d_out, int out_size, void* d_ws, size_t ws_size,
                              hipStream_t stream) {
    const float* x    = (const float*)d_in[0];   // (4096, 8, 1024)
    const float* h0   = (const float*)d_in[1];   // (8, 256)
    const float* Wi2h = (const float*)d_in[2];   // (1280, 256)
    const float* bi2h = (const float*)d_in[3];   // (256,)
    const float* Wh2o = (const float*)d_in[4];   // (256, 128)
    const float* bh2o = (const float*)d_in[5];   // (128,)
    float* out = (float*)d_out;                  // (4096, 8, 128)

    char* ws = (char*)d_ws;
    const size_t offWxh  = 0;                              // 1024*256 f16
    const size_t offWhh  = offWxh + 262144u * 2;           // 256*256 f16
    const size_t offWh2o = offWhh + 65536u * 2;            // 256*128 f16
    const size_t offXp   = offWh2o + 32768u * 2;           // 32768*256 f32
    const size_t offHs   = offXp + (size_t)MROWS * HIDDEN * 4;  // 32768*256 f16

    _Float16* WxhSw  = (_Float16*)(ws + offWxh);
    _Float16* WhhSw  = (_Float16*)(ws + offWhh);
    _Float16* Wh2oSw = (_Float16*)(ws + offWh2o);
    float*    xpbuf  = (float*)(ws + offXp);
    _Float16* hsbuf  = (_Float16*)(ws + offHs);

    // one-time weight convert + fragment-order swizzle
    rnn_swizzle_w<<<1024, 256, 0, stream>>>(Wi2h, WxhSw, INPUT, HIDDEN, 5);
    rnn_swizzle_w<<<256, 256, 0, stream>>>(Wi2h + (size_t)INPUT * HIDDEN, WhhSw,
                                           HIDDEN, HIDDEN, 3);
    rnn_swizzle_w<<<128, 256, 0, stream>>>(Wh2o, Wh2oSw, HIDDEN, OUTPUT, 3);

    // x_proj = x @ W_xh + b   (parallel GEMM, WMMA f16->f32)
    rnn_gemm_xproj<<<dim3(MROWS / 64, HIDDEN / 16), 128, 0, stream>>>(
        x, WxhSw, bi2h, xpbuf);

    // sequential recurrence (single persistent workgroup)
    rnn_scan<<<1, 512, 0, stream>>>(xpbuf, WhhSw, h0, hsbuf);

    // outputs = hs @ W_h2o + b   (parallel GEMM, WMMA f16->f32)
    rnn_gemm_out<<<dim3(MROWS / 64, OUTPUT / 16), 128, 0, stream>>>(
        hsbuf, Wh2oSw, bh2o, out);
}
